// RITPActionEncoder_23270132810285
// MI455X (gfx1250) — matrix operand order repaired and checked by reading, not compiled, not run on hardware
//
#include <hip/hip_runtime.h>
#include <hip/hip_bf16.h>

// ---------------------------------------------------------------------------
// Problem constants (from the reference)
// ---------------------------------------------------------------------------
#define B_      512
#define T_      50
#define NT2M_   30
#define FSTEPS_ 60
#define NA_     64
#define PPOLY_  512
#define H_      128
#define NHEAD_  8
#define HDIM_   16
#define NFREQ_  64
#define NLAYER_ 2
#define KPL_    128
#define KA_     48
#define PI_     3.14159265358979323846f
#define TWOPI_  6.28318530717958647692f

typedef __attribute__((ext_vector_type(16))) __bf16 v16bf;
typedef __attribute__((ext_vector_type(8)))  float  v8f;

// Padded A-tile row: 36 words = 144 B (16B-aligned for float4 LDS stores);
// 36*i mod 64 is distinct for i = 0..15 -> conflict-free fragment reads.
#define SA_PAD 36

// ---------------------------------------------------------------------------
// WMMA GEMM:  C[M,N] (+)= act( A[M,K] @ B[K,N] + bias )
//   - 1 block = 128 threads = 4 waves, each wave owns a 16x16 C tile
//     (block tile: 64 rows of M x 16 cols of N), K tiled by 32.
//   - A/B staged through LDS in f32 (float4 vectorized on the aligned fast
//     path), packed to bf16 fragments per ISA 7.12.2, accumulated in f32 with
//     v_wmma_f32_16x16x32_bf16.
//   - Template parameters keep the hot instantiations branch-free:
//       FOURIER : A row m, col k generated on the fly from x = fx[m*fxs]:
//                 k<64 -> cos(x*freqs[k]*2pi), k<128 -> sin(...), k==128 -> x
//       TRANSB  : B stored (N,K) row-major (GRU  x @ W.T)
//       GUARD   : ragged M/N/K bounds checks (only fourier K=129, critic N=1)
//   - Next K-tile is prefetched (global_prefetch_b8) while staging current.
// ---------------------------------------------------------------------------
template <bool FOURIER, bool TRANSB, bool GUARD>
__global__ __launch_bounds__(128)
void k_gemm_wmma(const float* __restrict__ A, int lda,
                 const float* __restrict__ Bm, int ldb,
                 const float* __restrict__ bias,
                 float* __restrict__ C, int ldc,
                 int M, int N, int K, int accumulate, int act,
                 const float* __restrict__ fx, int fxs,
                 const float* __restrict__ freqs)
{
  __shared__ float sA[64][SA_PAD];
  __shared__ float sB[32][16];
  const int tid  = threadIdx.x;
  const int wv   = tid >> 5;
  const int lane = tid & 31;
  const int half = lane >> 4;     // lane group 0..15 vs 16..31
  const int mrow = lane & 15;
  const int n0 = blockIdx.x * 16;
  const int m0 = blockIdx.y * 64;

  v8f acc = {};

  for (int k0 = 0; k0 < K; k0 += 32) {
    if constexpr (!FOURIER && !GUARD) {
      // ---- fast path: float4 staging (lda/ldb multiples of 4, tiles full) --
#pragma unroll
      for (int j = 0; j < 4; ++j) {
        const int idx = tid + j * 128;        // 0..511 quads of A tile
        const int mm = idx >> 3, kq = (idx & 7) * 4;
        const float4 v = *(const float4*)(A + (long)(m0 + mm) * lda + k0 + kq);
        *(float4*)&sA[mm][kq] = v;
      }
      if constexpr (!TRANSB) {
        const int kk = tid >> 2, nn = (tid & 3) * 4;
        const float4 v = *(const float4*)(Bm + (long)(k0 + kk) * ldb + n0 + nn);
        sB[kk][nn + 0] = v.x; sB[kk][nn + 1] = v.y;
        sB[kk][nn + 2] = v.z; sB[kk][nn + 3] = v.w;
      } else {
        const int nn = tid & 15, kq = (tid >> 4) * 4;
        const float4 v = *(const float4*)(Bm + (long)(n0 + nn) * ldb + k0 + kq);
        sB[kq + 0][nn] = v.x; sB[kq + 1][nn] = v.y;
        sB[kq + 2][nn] = v.z; sB[kq + 3][nn] = v.w;
      }
    } else {
      // ---- guarded / generated-A path (fourier K=129, critic N=1) ----
      for (int i = tid; i < 64 * 32; i += 128) {
        const int mm = i >> 5, kk = i & 31;
        const int m = m0 + mm, k = k0 + kk;
        float v = 0.f;
        if constexpr (FOURIER) {
          if (m < M && k < K) {
            const float x = fx[(long)m * fxs];
            if (k < NFREQ_)            v = __cosf(x * freqs[k] * TWOPI_);
            else if (k < 2 * NFREQ_)   v = __sinf(x * freqs[k - NFREQ_] * TWOPI_);
            else                       v = x;
          }
        } else {
          if (m < M && k < K) v = A[(long)m * lda + k];
        }
        sA[mm][kk] = v;
      }
      for (int i = tid; i < 32 * 16; i += 128) {
        const int kk = i >> 4, nn = i & 15;
        const int k = k0 + kk, n = n0 + nn;
        float v = 0.f;
        if (k < K && n < N)
          v = TRANSB ? Bm[(long)n * ldb + k] : Bm[(long)k * ldb + n];
        sB[kk][nn] = v;
      }
    }

    // ---- prefetch next K-tile while this one is consumed ----
    if constexpr (!FOURIER) {
      const int kn = k0 + 32;
      if (kn < K) {
        if (tid < 64) {
          const int m = m0 + tid;
          if (!GUARD || m < M)
            __builtin_prefetch(A + (long)m * lda + kn, 0, 1);
        } else if (tid < 96) {
          const int kk = tid - 64;
          if (!GUARD || kn + kk < K) {
            const float* bp = TRANSB ? (Bm + (long)(n0 + (kk & 15)) * ldb + kn)
                                     : (Bm + (long)(kn + kk) * ldb + n0);
            __builtin_prefetch(bp, 0, 1);
          }
        }
      }
    }
    __syncthreads();

    // ---- pack fragments (ISA 7.12.2 layouts) and issue WMMA ----
    v16bf af, bf;
#pragma unroll
    for (int i = 0; i < 16; ++i) {
      // A 16x32 (MxK): lanes0-15 vgprs hold K 0-7 then 16-23; lanes16-31: 8-15 then 24-31
      const int ka = ((i < 8) ? i : (i + 8)) + (half ? 8 : 0);
      af[i] = (__bf16)sA[wv * 16 + mrow][ka];
      // B 32x16 (KxN): lane = N; same K order as A: lanes0-15 K 0-7 then 16-23; lanes16-31: 8-15 then 24-31
      const int kb = ((i < 8) ? i : (i + 8)) + (half ? 8 : 0);
      bf[i] = (__bf16)sB[kb][mrow];
    }
    acc = __builtin_amdgcn_wmma_f32_16x16x32_bf16(
        false, af, false, bf, (short)0, acc, false, false);
    __syncthreads();
  }

  // ---- write C: vgpr r -> M = r (lanes 0-15) / 8+r (lanes 16-31), N = lane%16
  const int mb = m0 + wv * 16 + (half ? 8 : 0);
#pragma unroll
  for (int r = 0; r < 8; ++r) {
    const int m = mb + r, n = n0 + mrow;
    if (!GUARD || (m < M && n < N)) {
      float v = acc[r];
      if (bias)       v += bias[n];
      if (accumulate) v += C[(long)m * ldc + n];
      if (act == 1)   v = fmaxf(v, 0.f);
      C[(long)m * ldc + n] = v;
    }
  }
}

// ---------------------------------------------------------------------------
// LayerNorm over H=128, one wave per row (wave32, 4 floats/lane).
//   gidx  : optional gather (input row = gidx[row])  -> LN(x_src[src[e]])
//   resid : optional residual (out = resid + LN(in))
// ---------------------------------------------------------------------------
__global__ __launch_bounds__(32)
void k_layernorm(const float* __restrict__ in, const int* __restrict__ gidx,
                 float* __restrict__ out,
                 const float* __restrict__ gamma, const float* __restrict__ beta,
                 const float* __restrict__ resid, int relu, int rows)
{
  int row = blockIdx.x;
  if (row >= rows) return;
  long irow = gidx ? (long)gidx[row] : (long)row;
  const float4 v = ((const float4*)(in + irow * H_))[threadIdx.x];
  float s = v.x + v.y + v.z + v.w;
#pragma unroll
  for (int o = 16; o > 0; o >>= 1) s += __shfl_xor(s, o, 32);
  const float mean = s * (1.f / H_);
  float4 d = make_float4(v.x - mean, v.y - mean, v.z - mean, v.w - mean);
  float ss = d.x * d.x + d.y * d.y + d.z * d.z + d.w * d.w;
#pragma unroll
  for (int o = 16; o > 0; o >>= 1) ss += __shfl_xor(ss, o, 32);
  const float rstd = rsqrtf(ss * (1.f / H_) + 1e-5f);
  const float4 g = ((const float4*)gamma)[threadIdx.x];
  const float4 b = ((const float4*)beta)[threadIdx.x];
  float4 y = make_float4(d.x * rstd * g.x + b.x, d.y * rstd * g.y + b.y,
                         d.z * rstd * g.z + b.z, d.w * rstd * g.w + b.w);
  if (resid) {
    const float4 r4 = ((const float4*)(resid + (long)row * H_))[threadIdx.x];
    y.x += r4.x; y.y += r4.y; y.z += r4.z; y.w += r4.w;
  }
  if (relu) {
    y.x = fmaxf(y.x, 0.f); y.y = fmaxf(y.y, 0.f);
    y.z = fmaxf(y.z, 0.f); y.w = fmaxf(y.w, 0.f);
  }
  ((float4*)(out + (long)row * H_))[threadIdx.x] = y;
}

// ---------------------------------------------------------------------------
// Per-edge relative features: [norm, ang, rel_heading (, tfeat)]
// pos_m/head_m come from pos_t[:, T-1] / head_t[:, T-1].
// ---------------------------------------------------------------------------
__device__ __forceinline__ float wrapf(float a)
{
  float m = fmodf(a + PI_, TWOPI_);
  if (m < 0.f) m += TWOPI_;
  return m - PI_;
}

__global__ void k_rel_feats(const float* __restrict__ p_src,
                            const float* __restrict__ h_src,
                            const float* __restrict__ pos_t,
                            const float* __restrict__ head_t,
                            const int* __restrict__ src,
                            const int* __restrict__ dst,
                            float* __restrict__ feat, int E, int dd, int withT)
{
  int e = blockIdx.x * blockDim.x + threadIdx.x;
  if (e >= E) return;
  int s = src[e], d = dst[e];
  float pmx = pos_t[((long)d * T_ + (T_ - 1)) * 2 + 0];
  float pmy = pos_t[((long)d * T_ + (T_ - 1)) * 2 + 1];
  float hm  = head_t[(long)d * T_ + (T_ - 1)];
  float rpx = p_src[(long)s * 2 + 0] - pmx;
  float rpy = p_src[(long)s * 2 + 1] - pmy;
  float rh  = wrapf(h_src[s] - hm);
  float cvx = __cosf(hm), cvy = __sinf(hm);
  float ang = atan2f(cvx * rpy - cvy * rpx, cvx * rpx + cvy * rpy);
  float nrm = sqrtf(rpx * rpx + rpy * rpy);
  float* f = feat + (long)e * dd;
  f[0] = nrm; f[1] = ang; f[2] = rh;
  if (withT) f[3] = (float)(s % T_) - (float)(T_ - 1);
}

// ---------------------------------------------------------------------------
// GRU gate:  h = (1-z)*n + z*h   (gi precomputed for all steps)
// gi row for (b, f) = b*FSTEPS + f, width 3H = [ir | iz | in]
// ---------------------------------------------------------------------------
__global__ void k_gru_gate(const float* __restrict__ gi_all,
                           const float* __restrict__ gh,
                           float* __restrict__ h, int f)
{
  int i = blockIdx.x * blockDim.x + threadIdx.x;
  if (i >= B_ * H_) return;
  int b = i >> 7, c = i & 127;
  const float* gib = gi_all + ((long)b * FSTEPS_ + f) * (3 * H_);
  const float* ghb = gh + (long)b * (3 * H_);
  float r = 1.f / (1.f + __expf(-(gib[c]          + ghb[c])));
  float z = 1.f / (1.f + __expf(-(gib[H_ + c]     + ghb[H_ + c])));
  float n = tanhf(gib[2 * H_ + c] + r * ghb[2 * H_ + c]);
  h[i] = (1.f - z) * n + z * h[i];
}

__global__ void k_bcast_row(const float* __restrict__ v, float* __restrict__ out, int n)
{
  int i = blockIdx.x * blockDim.x + threadIdx.x;
  if (i < n) out[i] = v[i & (H_ - 1)];
}

// upd = agg + sigmoid(gpre) * (s - agg)
__global__ void k_gate_upd(const float* __restrict__ agg, const float* __restrict__ gpre,
                           const float* __restrict__ s, float* __restrict__ upd, int n)
{
  int i = blockIdx.x * blockDim.x + threadIdx.x;
  if (i >= n) return;
  float a = agg[i];
  float g = 1.f / (1.f + __expf(-gpre[i]));
  upd[i] = a + g * (s[i] - a);
}

// ---------------------------------------------------------------------------
// Attention softmax + aggregate. Edges for dst b are rows [b*Epd, (b+1)*Epd)
// (dst = repeat(arange(B), Epd) in the reference). 8 heads x HDIM 16.
// One block (128 threads) per destination.
// ---------------------------------------------------------------------------
__global__ __launch_bounds__(128)
void k_attn_agg(const float* __restrict__ q,   // (B, 128)
                const float* __restrict__ kj,  // (E, 128)
                const float* __restrict__ vj,  // (E, 128)
                float* __restrict__ agg,       // (B, 128)
                int Epd)
{
  __shared__ float sq[H_];
  __shared__ float ssim[KPL_ * NHEAD_];   // max Epd=128 -> 4KB
  __shared__ float sden[NHEAD_];
  const int b = blockIdx.x, tid = threadIdx.x;
  sq[tid] = q[(long)b * H_ + tid];
  __syncthreads();

  for (int i = tid; i < Epd * NHEAD_; i += 128) {
    int e = i >> 3, hh = i & 7;
    const float* kr = kj + ((long)b * Epd + e) * H_ + hh * HDIM_;
    float s = 0.f;
#pragma unroll
    for (int t = 0; t < HDIM_; ++t) s += sq[hh * HDIM_ + t] * kr[t];
    ssim[e * NHEAD_ + hh] = s * 0.25f;          // HDIM^-0.5
  }
  __syncthreads();

  if (tid < NHEAD_) {
    float mx = -3.4e38f;
    for (int e = 0; e < Epd; ++e) mx = fmaxf(mx, ssim[e * NHEAD_ + tid]);
    float s = 0.f;
    for (int e = 0; e < Epd; ++e) {
      float ex = __expf(ssim[e * NHEAD_ + tid] - mx);
      ssim[e * NHEAD_ + tid] = ex;
      s += ex;
    }
    sden[tid] = s;
  }
  __syncthreads();

  const int hh = tid >> 4;                       // head of this output channel
  const float inv = 1.f / sden[hh];
  float a = 0.f;
  for (int e = 0; e < Epd; ++e)
    a += ssim[e * NHEAD_ + hh] * vj[((long)b * Epd + e) * H_ + tid];
  agg[(long)b * H_ + tid] = a * inv;
}

// ---------------------------------------------------------------------------
// Host-side param tables (pytree-flattened: dict keys sorted, lists in order)
// ---------------------------------------------------------------------------
struct FourierP {
  const float *W1, *W2, *Wo, *b1, *b2, *be1, *beo, *bo, *freqs, *g1, *go;
};
struct AttnP {
  const float *Wf1, *Wf2, *Wg, *Wk, *Wkr, *Wout, *Wq, *Ws, *Wv, *Wvr;
  const float *b_dst, *b_fpost, *b_fpre, *b_post, *b_r, *b_src;
  const float *bf1, *bf2, *bg, *bout, *bq, *bs, *bv, *bvr;
  const float *g_dst, *g_fpost, *g_fpre, *g_post, *g_r, *g_src;
};

extern "C" void kernel_launch(void* const* d_in, const int* in_sizes, int n_in,
                              void* d_out, int out_size, void* d_ws, size_t ws_size,
                              hipStream_t stream)
{
  (void)in_sizes; (void)n_in; (void)out_size; (void)ws_size;

  // ---- top-level inputs (setup_inputs() dict order) ----
  const float* action    = (const float*)d_in[0];
  const float* pos_t     = (const float*)d_in[1];
  const float* head_t    = (const float*)d_in[2];
  const float* pos_a     = (const float*)d_in[3];
  const float* head_a    = (const float*)d_in[4];
  const float* pos_pl    = (const float*)d_in[5];
  const float* orient_pl = (const float*)d_in[6];
  const float* x_t       = (const float*)d_in[7];
  const float* x_pl      = (const float*)d_in[8];
  const float* x_a       = (const float*)d_in[9];
  const int*   e_t2m     = (const int*)d_in[10];
  const int*   e_pl2m    = (const int*)d_in[11];
  const int*   e_a2m     = (const int*)d_in[12];

  const int E1 = B_ * NT2M_, E2 = B_ * KPL_, E3 = B_ * KA_;
  const int* s_t2m  = e_t2m;        const int* d_t2m  = e_t2m  + E1;
  const int* s_pl2m = e_pl2m;       const int* d_pl2m = e_pl2m + E2;
  const int* s_a2m  = e_a2m;        const int* d_a2m  = e_a2m  + E3;

  // ---- params: leaves from d_in[13..] in pytree order (sorted dict keys) ----
  int pi = 13;
  auto nxt = [&]() { return (const float*)d_in[pi++]; };
  auto loadAttn = [&](AttnP& p) {
    p.Wf1 = nxt(); p.Wf2 = nxt(); p.Wg = nxt(); p.Wk = nxt(); p.Wkr = nxt();
    p.Wout = nxt(); p.Wq = nxt(); p.Ws = nxt(); p.Wv = nxt(); p.Wvr = nxt();
    p.b_dst = nxt(); p.b_fpost = nxt(); p.b_fpre = nxt(); p.b_post = nxt();
    p.b_r = nxt(); p.b_src = nxt(); p.bf1 = nxt(); p.bf2 = nxt(); p.bg = nxt();
    p.bout = nxt(); p.bq = nxt(); p.bs = nxt(); p.bv = nxt(); p.bvr = nxt();
    p.g_dst = nxt(); p.g_fpost = nxt(); p.g_fpre = nxt(); p.g_post = nxt();
    p.g_r = nxt(); p.g_src = nxt();
  };
  auto loadFour = [&](FourierP& p) {
    p.W1 = nxt(); p.W2 = nxt(); p.Wo = nxt(); p.b1 = nxt(); p.b2 = nxt();
    p.be1 = nxt(); p.beo = nxt(); p.bo = nxt(); p.freqs = nxt();
    p.g1 = nxt(); p.go = nxt();
  };

  AttnP a2m[2], pl2m[2], t2m[2];
  FourierP fr_a2m, fr_pl2m, fr_t2m, fy;
  loadAttn(a2m[0]); loadAttn(a2m[1]);
  const float *cW1 = nxt(), *cW2 = nxt(), *cb1 = nxt(), *cb2 = nxt(),
              *cbe = nxt(), *cg = nxt();
  const float *gWh = nxt(), *gWi = nxt(), *gbh = nxt(), *gbi = nxt();
  const float *h0 = nxt();
  loadAttn(pl2m[0]); loadAttn(pl2m[1]);
  loadFour(fr_a2m); loadFour(fr_pl2m); loadFour(fr_t2m);
  loadAttn(t2m[0]); loadAttn(t2m[1]);
  loadFour(fy);

  // ---- workspace bump allocator ----
  char* wsp = (char*)d_ws;
  size_t off = 0;
  auto alloc = [&](size_t nfl) {
    float* p = (float*)(wsp + off);
    off += ((nfl * sizeof(float) + 255) / 256) * 256;
    return p;
  };
  const int EMAX = E2;                       // 65536
  float* feat = alloc((size_t)EMAX * 4);
  float* r1   = alloc((size_t)E1 * H_);
  float* r2   = alloc((size_t)E2 * H_);
  float* r3   = alloc((size_t)E3 * H_);
  float* yall = alloc((size_t)B_ * FSTEPS_ * H_);
  float* tb   = alloc((size_t)EMAX * H_);    // fourier t / attn xs-gather
  float* hb   = alloc((size_t)EMAX * H_);    // fourier h / attn rn
  float* kj   = alloc((size_t)EMAX * H_);
  float* vj   = alloc((size_t)EMAX * H_);    // contiguous after kj
  float* gi   = kj;                          // GRU gi aliases kj+vj (disjoint lifetime)
  float* ghb  = alloc((size_t)B_ * 3 * H_);
  float* mbuf = alloc((size_t)B_ * H_);
  float* xd   = alloc((size_t)B_ * H_);
  float* qb   = alloc((size_t)B_ * H_);
  float* aggb = alloc((size_t)B_ * H_);
  float* gp   = alloc((size_t)B_ * H_);
  float* sb   = alloc((size_t)B_ * H_);
  float* ub   = alloc((size_t)B_ * H_);
  float* ob   = alloc((size_t)B_ * H_);
  float* xln  = alloc((size_t)B_ * H_);
  float* ff1  = alloc((size_t)B_ * 4 * H_);
  float* c1   = alloc((size_t)B_ * H_);

  // ---- launch helpers ----
  // Plain GEMM; picks the unguarded fast path when fully tile-aligned.
  auto gemm = [&](const float* A, int lda, const float* Bm, int ldb, int trB,
                  const float* bias, float* C, int ldc, int M, int N, int K,
                  int acc, int act) {
    dim3 g((N + 15) / 16, (M + 63) / 64);
    const bool full = (M % 64 == 0) && (N % 16 == 0) && (K % 32 == 0);
    if (trB) {
      if (full)
        k_gemm_wmma<false, true, false><<<g, 128, 0, stream>>>(
            A, lda, Bm, ldb, bias, C, ldc, M, N, K, acc, act, nullptr, 0, nullptr);
      else
        k_gemm_wmma<false, true, true><<<g, 128, 0, stream>>>(
            A, lda, Bm, ldb, bias, C, ldc, M, N, K, acc, act, nullptr, 0, nullptr);
    } else {
      if (full)
        k_gemm_wmma<false, false, false><<<g, 128, 0, stream>>>(
            A, lda, Bm, ldb, bias, C, ldc, M, N, K, acc, act, nullptr, 0, nullptr);
      else
        k_gemm_wmma<false, false, true><<<g, 128, 0, stream>>>(
            A, lda, Bm, ldb, bias, C, ldc, M, N, K, acc, act, nullptr, 0, nullptr);
    }
  };
  // Fourier-feature GEMM (K = 129, always guarded).
  auto gemm_fourier = [&](const float* Bm, const float* bias, float* C,
                          int M, int N, int K,
                          const float* fx, int fxs, const float* freqs) {
    dim3 g((N + 15) / 16, (M + 63) / 64);
    k_gemm_wmma<true, false, true><<<g, 128, 0, stream>>>(
        nullptr, 0, Bm, H_, bias, C, H_, M, N, K, 0, 0, fx, fxs, freqs);
  };
  auto ln = [&](const float* in, const int* gidx, float* out, const float* g,
                const float* b, const float* resid, int relu, int rows) {
    k_layernorm<<<rows, 32, 0, stream>>>(in, gidx, out, g, b, resid, relu, rows);
  };

  // ---- Fourier encoder: r = relu(LN(sum_d relu(LN(feat_d@W1_d+b1_d))@W2_d+b2_d)) @ Wo + bo
  auto run_fourier = [&](const FourierP& p, const float* fx, int dd, int E, float* rout) {
    for (int d = 0; d < dd; ++d) {
      gemm_fourier(p.W1 + (long)d * (2 * NFREQ_ + 1) * H_, p.b1 + d * H_,
                   tb, E, H_, 2 * NFREQ_ + 1, fx + d, dd, p.freqs + d * NFREQ_);
      ln(tb, nullptr, tb, p.g1 + d * H_, p.be1 + d * H_, nullptr, 1, E);
      gemm(tb, H_, p.W2 + (long)d * H_ * H_, H_, 0, p.b2 + d * H_,
           hb, H_, E, H_, H_, (d > 0), 0);
    }
    ln(hb, nullptr, hb, p.go, p.beo, nullptr, 1, E);
    gemm(hb, H_, p.Wo, H_, 0, p.bo, rout, H_, E, H_, H_, 0, 0);
  };

  // ---- 1. relative edge features + Fourier relation encoders ----
  k_rel_feats<<<(E1 + 255) / 256, 256, 0, stream>>>(pos_t, head_t, pos_t, head_t,
                                                    s_t2m, d_t2m, feat, E1, 4, 1);
  run_fourier(fr_t2m, feat, 4, E1, r1);
  k_rel_feats<<<(E2 + 255) / 256, 256, 0, stream>>>(pos_pl, orient_pl, pos_t, head_t,
                                                    s_pl2m, d_pl2m, feat, E2, 3, 0);
  run_fourier(fr_pl2m, feat, 3, E2, r2);
  k_rel_feats<<<(E3 + 255) / 256, 256, 0, stream>>>(pos_a, head_a, pos_t, head_t,
                                                    s_a2m, d_a2m, feat, E3, 3, 0);
  run_fourier(fr_a2m, feat, 3, E3, r3);
  // action encoder: x = action reshaped (B*FSTEPS, 2)
  run_fourier(fy, action, 2, B_ * FSTEPS_, yall);

  // ---- 2. GRU over FSTEPS (gi hoisted into one GEMM; gh per step) ----
  gemm(yall, H_, gWi, H_, /*transB=*/1, gbi, gi, 3 * H_,
       B_ * FSTEPS_, 3 * H_, H_, 0, 0);
  k_bcast_row<<<(B_ * H_ + 255) / 256, 256, 0, stream>>>(h0, mbuf, B_ * H_);
  for (int f = 0; f < FSTEPS_; ++f) {
    gemm(mbuf, H_, gWh, H_, /*transB=*/1, gbh, ghb, 3 * H_, B_, 3 * H_, H_, 0, 0);
    k_gru_gate<<<(B_ * H_ + 255) / 256, 256, 0, stream>>>(gi, ghb, mbuf, f);
  }

  // ---- 3. relational attention layers ----
  auto run_attn = [&](const AttnP& p, const float* x_src, const float* r,
                      const int* src, int Epd, int E) {
    float* xsg = tb;   // gathered+LN'd source rows (every edge has a unique src)
    float* rn  = hb;
    ln(mbuf, nullptr, xd, p.g_dst, p.b_dst, nullptr, 0, B_);
    ln(x_src, src, xsg, p.g_src, p.b_src, nullptr, 0, E);
    ln(r, nullptr, rn, p.g_r, p.b_r, nullptr, 0, E);
    gemm(xd, H_, p.Wq, H_, 0, p.bq, qb, H_, B_, H_, H_, 0, 0);
    gemm(xsg, H_, p.Wk,  H_, 0, nullptr, kj, H_, E, H_, H_, 0, 0);
    gemm(rn,  H_, p.Wkr, H_, 0, nullptr, kj, H_, E, H_, H_, 1, 0); // kj = k[src]+kr
    gemm(xsg, H_, p.Wv,  H_, 0, p.bv,  vj, H_, E, H_, H_, 0, 0);
    gemm(rn,  H_, p.Wvr, H_, 0, p.bvr, vj, H_, E, H_, H_, 1, 0);   // vj = v[src]+vr
    k_attn_agg<<<B_, 128, 0, stream>>>(qb, kj, vj, aggb, Epd);
    // g = sigmoid([agg|xd] @ Wg + bg): split Wg rows, accumulate
    gemm(aggb, H_, p.Wg,             H_, 0, nullptr, gp, H_, B_, H_, H_, 0, 0);
    gemm(xd,   H_, p.Wg + H_ * H_,   H_, 0, p.bg,    gp, H_, B_, H_, H_, 1, 0);
    gemm(xd, H_, p.Ws, H_, 0, p.bs, sb, H_, B_, H_, H_, 0, 0);
    k_gate_upd<<<(B_ * H_ + 255) / 256, 256, 0, stream>>>(aggb, gp, sb, ub, B_ * H_);
    gemm(ub, H_, p.Wout, H_, 0, p.bout, ob, H_, B_, H_, H_, 0, 0);
    ln(ob, nullptr, mbuf, p.g_post, p.b_post, /*resid=*/mbuf, 0, B_); // x = x + LN(out)
    ln(mbuf, nullptr, xln, p.g_fpre, p.b_fpre, nullptr, 0, B_);
    gemm(xln, H_, p.Wf1, 4 * H_, 0, p.bf1, ff1, 4 * H_, B_, 4 * H_, H_, 0, /*relu=*/1);
    gemm(ff1, 4 * H_, p.Wf2, H_, 0, p.bf2, ob, H_, B_, H_, 4 * H_, 0, 0);
    ln(ob, nullptr, mbuf, p.g_fpost, p.b_fpost, /*resid=*/mbuf, 0, B_);
  };

  for (int i = 0; i < NLAYER_; ++i) {
    run_attn(t2m[i],  x_t,  r1, s_t2m,  NT2M_, E1);
    run_attn(pl2m[i], x_pl, r2, s_pl2m, KPL_,  E2);
    run_attn(a2m[i],  x_a,  r3, s_a2m,  KA_,   E3);
  }

  // ---- 4. critic head ----
  gemm(mbuf, H_, cW1, H_, 0, cb1, c1, H_, B_, H_, H_, 0, 0);
  ln(c1, nullptr, c1, cg, cbe, nullptr, 1, B_);
  gemm(c1, H_, cW2, 1, 0, cb2, (float*)d_out, 1, B_, 1, H_, 0, 0);
}
